// Readout_5746666242200
// MI455X (gfx1250) — compile-verified, matrix operand order checked by reading, not run on hardware
//
#include <hip/hip_runtime.h>
#include <hip/hip_bf16.h>
#include <math.h>

typedef __attribute__((ext_vector_type(16))) __bf16 v16bf;
typedef __attribute__((ext_vector_type(8)))  float  v8f;
typedef __attribute__((ext_vector_type(4)))  float  f4;

#define DIM 128
#define SEG_CAP (1u << 20)

// workspace layout (bytes)
#define RBH_OFF   0          // 128x128 bf16, swizzled to WMMA B-fragment order
#define RBL_OFF   32768      // low-order bf16 residual (contiguous with hi)
#define CUB_OFF   65536      // 128 f32 combined bias (cu | cv)
#define THETA_OFF 66048      // 64 f32 rope thetas
#define SEG_OFF   131072     // SEG_CAP int32 segment-start table (4MB)
#define POS_OFF   (131072 + 4194304)   // N int32 positions (-1 = not roped)

// Map matrix element (kt,jt tile; kk in 0..31, n in 0..15) to the flat
// position a lane's v16bf B-fragment expects for V_WMMA_F32_16X16X32_BF16.
__device__ __forceinline__ int swz_index(int kt, int jt, int kk, int n) {
  int pair  = kk & 1;
  int u     = kk >> 1;                       // 0..15
  int r     = ((u >> 3) << 2) | (u & 3);     // VGPR index 0..7
  int khalf = (u >> 2) & 1;                  // lane half
  int lane  = n + (khalf << 4);
  int elem  = (r << 1) | pair;
  return ((jt * 4 + kt) * 32 + lane) * 16 + elem;
}

__global__ void precompute_kernel(const float* __restrict__ W1,
                                  const float* __restrict__ b1,
                                  const float* __restrict__ W2,
                                  unsigned short* __restrict__ rbh,
                                  unsigned short* __restrict__ rbl,
                                  float* __restrict__ cub,
                                  float* __restrict__ theta) {
  int c = threadIdx.x;            // 0..127 output column
  int i = c & 63;
  bool isU = (c < 64);
  float w2a = W2[2 * i], w2b = W2[2 * i + 1];
  for (int k = 0; k < DIM; ++k) {
    float a = W1[(2 * i) * DIM + k];
    float b = W1[(2 * i + 1) * DIM + k];
    float r = isU ? (a * w2a + b * w2b) : (a * w2b - b * w2a);
    __bf16 hi = (__bf16)r;
    __bf16 lo = (__bf16)(r - (float)hi);
    int idx = swz_index(k >> 5, c >> 4, k & 31, c & 15);
    rbh[idx] = __builtin_bit_cast(unsigned short, hi);
    rbl[idx] = __builtin_bit_cast(unsigned short, lo);
  }
  float ba = b1[2 * i], bb = b1[2 * i + 1];
  cub[c] = isU ? (ba * w2a + bb * w2b) : (ba * w2b - bb * w2a);
  if (c < 64)
    theta[c] = __expf(-(float)c * (9.210340371976184f / 64.0f)); // 10000^(-2c/128)
}

// Pass A: scatter run starts of the sorted batch ids into a small table.
__global__ void segstart_kernel(const long long* __restrict__ batch,
                                int* __restrict__ seg, int N) {
  int n = blockIdx.x * blockDim.x + threadIdx.x;
  if (n >= N) return;
  long long b = batch[n];
  if (n == 0 || batch[n - 1] != b) {
    unsigned id = (unsigned)b;
    if (id < SEG_CAP) seg[id] = n;
  }
}

// Pass B: per-row rope position; -1 marks the (un-roped) last segment.
__global__ void pos_kernel(const long long* __restrict__ batch,
                           const int* __restrict__ seg,
                           int* __restrict__ pos, int N) {
  int n = blockIdx.x * blockDim.x + threadIdx.x;
  if (n >= N) return;
  long long b = batch[n];
  long long lastb = batch[N - 1];
  pos[n] = (b == lastb) ? -1 : (n - seg[(unsigned)b]);
}

// CDNA5 async global->LDS copy (ASYNCcnt-tracked), bypasses VGPRs.
__device__ __forceinline__ void async_load_b128(unsigned lds_off, const void* gptr) {
  asm volatile("global_load_async_to_lds_b128 %0, %1, off"
               :: "v"(lds_off), "v"(gptr) : "memory");
}
__device__ __forceinline__ void wait_asynccnt0() {
  asm volatile("s_wait_asynccnt 0x0" ::: "memory");
}

__global__ void __launch_bounds__(256)
fused_gemm_rope_kernel(const float* __restrict__ x,
                       const int* __restrict__ pos_arr,
                       const unsigned short* __restrict__ rb,   // hi(32KB)+lo(32KB)
                       const float* __restrict__ cub,
                       const float* __restrict__ theta,
                       const float* __restrict__ b2p,
                       float* __restrict__ out,
                       int N, int ntiles) {
  __shared__ __align__(16) unsigned short Bsh[32768];  // 64KB: hi | lo

  const int tid   = threadIdx.x;
  const int lane  = tid & 31;
  const int wave  = tid >> 5;
  const int n     = lane & 15;
  const int khalf = lane >> 4;

  // ---- Stage the swizzled B matrix (hi+lo) into LDS with async copies ----
  {
    unsigned ldsbase = (unsigned)(size_t)(&Bsh[0]);
#pragma unroll
    for (int c = 0; c < 16; ++c) {
      int chunk = c * 256 + tid;            // 4096 x 16B chunks, coalesced
      async_load_b128(ldsbase + chunk * 16,
                      (const void*)((const char*)rb + chunk * 16));
    }
    wait_asynccnt0();
  }
  __syncthreads();

  // Per-lane loop-invariant epilogue constants (col = jt*16 + n)
  float cubv[8], thv[4];
#pragma unroll
  for (int jt = 0; jt < 8; ++jt) cubv[jt] = cub[jt * 16 + n];
#pragma unroll
  for (int jt = 0; jt < 4; ++jt) thv[jt] = theta[jt * 16 + n];
  const float b2 = b2p[0];

  // Each wave owns 32 rows (two 16-row strips) and all 128 columns.
  for (int tile = blockIdx.x; tile < ntiles; tile += gridDim.x) {
    const int rowbase = tile * 256 + wave * 32;

    int r0 = rowbase + n;      if (r0 > N - 1) r0 = N - 1;
    int r1 = rowbase + 16 + n; if (r1 > N - 1) r1 = N - 1;
    const f4* xrow0 = (const f4*)(x + (size_t)r0 * DIM);
    const f4* xrow1 = (const f4*)(x + (size_t)r1 * DIM);

    // One coalesced 4B load per row replaces the old binary search.
    int pos;
    {
      int r = rowbase + lane; if (r > N - 1) r = N - 1;
      pos = pos_arr[r];
    }

    { // prefetch next tile's rows for this lane
      long long nrow = (long long)(tile + (long long)gridDim.x) * 256 + wave * 32 + n;
      if (nrow < N) __builtin_prefetch(x + (size_t)nrow * DIM, 0, 0);
    }

    v8f acc[2][8];
#pragma unroll
    for (int s = 0; s < 2; ++s)
#pragma unroll
      for (int jt = 0; jt < 8; ++jt) acc[s][jt] = (v8f){};

#pragma unroll
    for (int kt = 0; kt < 4; ++kt) {
      const int f0 = kt * 8 + khalf * 2;     // float4 units within row
      v16bf ah[2], al[2];
#pragma unroll
      for (int s = 0; s < 2; ++s) {
        const f4* xr = s ? xrow1 : xrow0;
        f4 q0 = __builtin_nontemporal_load(xr + f0);
        f4 q1 = __builtin_nontemporal_load(xr + f0 + 1);
        f4 q2 = __builtin_nontemporal_load(xr + f0 + 4);
        f4 q3 = __builtin_nontemporal_load(xr + f0 + 5);
        float vals[16] = {q0.x, q0.y, q0.z, q0.w, q1.x, q1.y, q1.z, q1.w,
                          q2.x, q2.y, q2.z, q2.w, q3.x, q3.y, q3.z, q3.w};
#pragma unroll
        for (int e = 0; e < 16; ++e) {
          float v = vals[e];
          __bf16 h = (__bf16)v;
          ah[s][e] = h;
          al[s][e] = (__bf16)(v - (float)h);
        }
      }
#pragma unroll
      for (int jt = 0; jt < 8; ++jt) {
        const int off16 = ((jt * 4 + kt) * 32 + lane) * 16;
        v16bf bh = *(const v16bf*)(const void*)(Bsh + off16);
        v16bf bl = *(const v16bf*)(const void*)(Bsh + 16384 + off16);
        // fp32-via-bf16 split: hi*hi + hi*lo + lo*hi, two row strips share B
        acc[0][jt] = __builtin_amdgcn_wmma_f32_16x16x32_bf16(false, ah[0], false, bh, (short)0, acc[0][jt], false, false);
        acc[0][jt] = __builtin_amdgcn_wmma_f32_16x16x32_bf16(false, ah[0], false, bl, (short)0, acc[0][jt], false, false);
        acc[0][jt] = __builtin_amdgcn_wmma_f32_16x16x32_bf16(false, al[0], false, bh, (short)0, acc[0][jt], false, false);
        acc[1][jt] = __builtin_amdgcn_wmma_f32_16x16x32_bf16(false, ah[1], false, bh, (short)0, acc[1][jt], false, false);
        acc[1][jt] = __builtin_amdgcn_wmma_f32_16x16x32_bf16(false, ah[1], false, bl, (short)0, acc[1][jt], false, false);
        acc[1][jt] = __builtin_amdgcn_wmma_f32_16x16x32_bf16(false, al[1], false, bh, (short)0, acc[1][jt], false, false);
      }
    }

    // ---- epilogue: out[row] = sum_jt w(row,col_jt) * (acc + cub) + b2 ----
#pragma unroll
    for (int s = 0; s < 2; ++s) {
#pragma unroll
      for (int r = 0; r < 8; ++r) {
        int srcl = s * 16 + khalf * 8 + r;       // lane holding this row's pos
        int p = __shfl(pos, srcl, 32);
        float sum;
        if (p < 0) {
          sum = (acc[s][0][r] + cubv[0]) + (acc[s][1][r] + cubv[1]) +
                (acc[s][2][r] + cubv[2]) + (acc[s][3][r] + cubv[3]);
        } else {
          float fp = (float)p;
          sum = 0.0f;
#pragma unroll
          for (int jt = 0; jt < 4; ++jt) {       // theta shared by cols c, c+64
            float sn, cs;
            __sincosf(fp * thv[jt], &sn, &cs);
            sum += cs * (acc[s][jt][r] + cubv[jt]);
            sum += sn * (acc[s][jt + 4][r] + cubv[jt + 4]);
          }
        }
        // reduce the 16 N-lanes of each half (masks 1..8 stay within halves)
        sum += __shfl_xor(sum, 1, 32);
        sum += __shfl_xor(sum, 2, 32);
        sum += __shfl_xor(sum, 4, 32);
        sum += __shfl_xor(sum, 8, 32);
        if (n == 0) {
          int orow = rowbase + s * 16 + khalf * 8 + r;
          if (orow < N) out[orow] = sum + b2;
        }
      }
    }
  }
}

extern "C" void kernel_launch(void* const* d_in, const int* in_sizes, int n_in,
                              void* d_out, int out_size, void* d_ws, size_t ws_size,
                              hipStream_t stream) {
  (void)n_in; (void)out_size; (void)ws_size;
  const float*     x     = (const float*)d_in[0];
  const long long* batch = (const long long*)d_in[1];
  const float*     W1    = (const float*)d_in[2];
  const float*     b1    = (const float*)d_in[3];
  const float*     W2    = (const float*)d_in[4];
  const float*     b2    = (const float*)d_in[5];
  float*           out   = (float*)d_out;

  const int N = in_sizes[0] / DIM;

  char* ws = (char*)d_ws;
  unsigned short* rbh   = (unsigned short*)(ws + RBH_OFF);
  unsigned short* rbl   = (unsigned short*)(ws + RBL_OFF);
  float*          cub   = (float*)(ws + CUB_OFF);
  float*          theta = (float*)(ws + THETA_OFF);
  int*            seg   = (int*)(ws + SEG_OFF);
  int*            pos   = (int*)(ws + POS_OFF);

  precompute_kernel<<<1, 128, 0, stream>>>(W1, b1, W2, rbh, rbl, cub, theta);

  const int nb = (N + 255) / 256;
  segstart_kernel<<<nb, 256, 0, stream>>>(batch, seg, N);
  pos_kernel<<<nb, 256, 0, stream>>>(batch, seg, pos, N);

  const int ntiles = (N + 255) / 256;            // 256 rows per block-pass
  int grid = ntiles < 768 ? ntiles : 768;        // amortize B staging per block
  fused_gemm_rope_kernel<<<grid, 256, 0, stream>>>(x, pos, rbh, cub,
                                                   theta, b2, out, N, ntiles);
}
